// Prior_23416161697849
// MI455X (gfx1250) — matrix operand order repaired and checked by reading, not compiled, not run on hardware
//
#include <hip/hip_runtime.h>
#include <hip/hip_bf16.h>
#include <math.h>

// ---------------------------------------------------------------------------
// Prior network pipeline for MI455X (gfx1250, wave32, WMMA).
//   GRU(T=25, A=16, H=128)  ->  LayerNorm  ->  GraphConv(E=800k) -> 4 heads
// All GEMMs run on V_WMMA_F32_16X16X32_F16 with f32 accumulation; the GRU
// hidden state stays resident in LDS (f16) for all 25 steps (rows are
// wave-private, so the time loop needs no barriers).
// ---------------------------------------------------------------------------

typedef __attribute__((ext_vector_type(16))) _Float16 v16h;
typedef __attribute__((ext_vector_type(8)))  _Float16 v8h;
typedef __attribute__((ext_vector_type(8)))  float    v8f;

#define T_STEPS 25
#define A_DIM   16
#define H_DIM   128
#define I_DIM   64

__device__ inline v8f wmma16(v16h a, v16h b, v8f c) {
  // (neg_a, A, neg_b, B, c_mod, C, reuse_a, reuse_b)
  return __builtin_amdgcn_wmma_f32_16x16x32_f16(false, a, false, b, (short)0, c,
                                                false, false);
}

// B operand: packed [chunk][tile][lane][16 halfs] -> one contiguous 32B/lane.
__device__ inline v16h loadB_g(const _Float16* __restrict__ base, int chunk,
                               int tile, int nTiles, int lane) {
  return *(const v16h*)(base + (((size_t)(chunk * nTiles + tile) * 32 + lane) << 4));
}

// A operand (16x32 f16, documented interleaved layout): lane half selects
// K base (0 or 8); halfs 0..7 = K kb..kb+7, halfs 8..15 = K kb+16..kb+23.
__device__ inline v16h loadA_lds(const _Float16* p) {
  v8h lo = *(const v8h*)(p);
  v8h hi = *(const v8h*)(p + 16);
  v16h a;
#pragma unroll
  for (int i = 0; i < 8; ++i) { a[i] = lo[i]; a[i + 8] = hi[i]; }
  return a;
}

__device__ inline v16h loadA_g32(const float* __restrict__ p, float scale) {
  v16h a;
#pragma unroll
  for (int i = 0; i < 8; ++i) {
    a[i]     = (_Float16)(p[i] * scale);
    a[i + 8] = (_Float16)(p[16 + i] * scale);
  }
  return a;
}

// ---------------------------------------------------------------------------
// Pack fp32 weight matrix into f16 WMMA B-operand layout:
// dst[((c*nTiles+g)*32+lane)*16+h] = B[k][n], n = g*16 + lane%16,
// k = c*32 + (lane>=16 ? 8 : 0) + h%8 + (h>=8 ? 16 : 0)  (interleaved B packing per ISA tables).
// wTrans=1: torch Linear layout W(out,in), B[k][n]=W[n*ldw+k].
// wTrans=0: (in,out) layout,              B[k][n]=W[k*ldw+n].
// ---------------------------------------------------------------------------
__global__ void pack_b_kernel(const float* __restrict__ W, _Float16* __restrict__ dstp,
                              int nChunks, int nTiles, int K, int Ncols,
                              int wTrans, int ldw) {
  int idx = blockIdx.x * 256 + threadIdx.x;
  int total = nChunks * nTiles * 512;
  if (idx >= total) return;
  int h    = idx & 15;
  int lane = (idx >> 4) & 31;
  int tg   = idx >> 9;
  int g    = tg % nTiles;
  int c    = tg / nTiles;
  int n = g * 16 + (lane & 15);
  int k = c * 32 + ((lane >> 4) << 3) + (h & 7) + ((h >> 3) << 4);
  float v = 0.f;
  if (k < K && n < Ncols)
    v = wTrans ? W[(size_t)n * ldw + k] : W[(size_t)k * ldw + n];
  dstp[idx] = (_Float16)v;
}

__global__ void zero_kernel(float* __restrict__ p, int n) {
  int i = blockIdx.x * 256 + threadIdx.x;
  if (i < n) p[i] = 0.f;
}

// ---------------------------------------------------------------------------
// GRU: 128 rows per block, 8 waves x 16-row strips; h resident in LDS (f16).
// Per step/wave: 8 j-tiles x (r:1+4, z:1+4, i_n:1, h_n:4) = 120 WMMAs.
// ---------------------------------------------------------------------------
__global__ __launch_bounds__(256)
void gru_kernel(const float* __restrict__ actions, const float* __restrict__ hidden,
                const _Float16* __restrict__ wih_pk, const _Float16* __restrict__ whh_pk,
                const float* __restrict__ b_ih, const float* __restrict__ b_hh,
                float* __restrict__ h_out, int Nn) {
  __shared__ _Float16 lds_h[128 * H_DIM];  // 32 KB
  const int lane = threadIdx.x & 31;
  const int wave = threadIdx.x >> 5;
  const int R = wave * 16;
  const size_t nbase = (size_t)blockIdx.x * 128;

  for (int i = threadIdx.x; i < 128 * H_DIM; i += 256) {
    int r = i >> 7, c = i & 127;
    size_t g = nbase + r; if (g >= (size_t)Nn) g = (size_t)Nn - 1;
    lds_h[i] = (_Float16)hidden[g * H_DIM + c];
  }
  __syncthreads();

  const int m = lane & 15;
  const int hsel = lane >> 4;
  const int lrow = R + m;
  size_t grow = nbase + lrow;
  if (grow >= (size_t)Nn) grow = (size_t)Nn - 1;
  const float* xbase = actions + grow * (T_STEPS * A_DIM) + hsel * 8;

  for (int t = 0; t < T_STEPS; ++t) {
    // x A-operand: K 0..15 valid, 16..31 zero-padded.
    v16h xa;
#pragma unroll
    for (int i = 0; i < 8; ++i) {
      xa[i]     = (_Float16)xbase[t * A_DIM + i];
      xa[i + 8] = (_Float16)0.0f;
    }
    // h A-operands for K = 0..127 (4 chunks of 32), wave-private rows.
    v16h ha[4];
#pragma unroll
    for (int c = 0; c < 4; ++c)
      ha[c] = loadA_lds(&lds_h[lrow * H_DIM + c * 32 + hsel * 8]);

    for (int j = 0; j < 8; ++j) {
      const int col = j * 16 + m;  // hidden-unit index (C-layout column)
      v8f accr = {};
      accr = wmma16(xa, loadB_g(wih_pk, 0, j, 24, lane), accr);
#pragma unroll
      for (int c = 0; c < 4; ++c)
        accr = wmma16(ha[c], loadB_g(whh_pk, c, j, 24, lane), accr);
      const float br = b_ih[col] + b_hh[col];

      v8f accz = {};
      accz = wmma16(xa, loadB_g(wih_pk, 0, 8 + j, 24, lane), accz);
#pragma unroll
      for (int c = 0; c < 4; ++c)
        accz = wmma16(ha[c], loadB_g(whh_pk, c, 8 + j, 24, lane), accz);
      const float bz = b_ih[H_DIM + col] + b_hh[H_DIM + col];

      v8f acci = {};
      acci = wmma16(xa, loadB_g(wih_pk, 0, 16 + j, 24, lane), acci);
      v8f acch = {};
#pragma unroll
      for (int c = 0; c < 4; ++c)
        acch = wmma16(ha[c], loadB_g(whh_pk, c, 16 + j, 24, lane), acch);
      const float bin = b_ih[2 * H_DIM + col];
      const float bhn = b_hh[2 * H_DIM + col];

#pragma unroll
      for (int r = 0; r < 8; ++r) {
        const int lr = R + r + hsel * 8;  // C-layout row
        float hold = (float)lds_h[lr * H_DIM + col];
        float rg = 1.f / (1.f + __expf(-(accr[r] + br)));
        float zg = 1.f / (1.f + __expf(-(accz[r] + bz)));
        float ng = tanhf(acci[r] + bin + rg * (acch[r] + bhn));
        float hn = (1.f - zg) * ng + zg * hold;
        lds_h[lr * H_DIM + col] = (_Float16)hn;
      }
    }
  }
  __syncthreads();
  for (int i = threadIdx.x; i < 128 * H_DIM; i += 256) {
    int r = i >> 7, c = i & 127;
    size_t g = nbase + r;
    if (g < (size_t)Nn) h_out[g * H_DIM + c] = (float)lds_h[i];
  }
}

// ---------------------------------------------------------------------------
// Graph pieces
// ---------------------------------------------------------------------------
__global__ void deg_kernel(const int* __restrict__ src, const int* __restrict__ dst,
                           float* __restrict__ dsrc, float* __restrict__ ddst, int E) {
  int e = blockIdx.x * 256 + threadIdx.x;
  if (e < E) {
    atomicAdd(&dsrc[src[e]], 1.f);
    atomicAdd(&ddst[dst[e]], 1.f);
  }
}

__global__ void norm_kernel(float* __restrict__ d, int Nn) {
  int n = blockIdx.x * 256 + threadIdx.x;
  if (n < Nn) { float x = d[n]; d[n] = rsqrtf(x > 0.f ? x : 1.f); }
}

// LayerNorm over H=128, one wave per row; folds the src-side graph norm.
__global__ __launch_bounds__(256)
void ln_kernel(const float* __restrict__ h, const float* __restrict__ g,
               const float* __restrict__ b, const float* __restrict__ nsrc,
               float* __restrict__ x_ln, int Nn) {
  int lane = threadIdx.x & 31;
  int row = blockIdx.x * 8 + (threadIdx.x >> 5);
  if (row >= Nn) return;
  const float* hp = h + (size_t)row * H_DIM + lane * 4;
  float v0 = hp[0], v1 = hp[1], v2 = hp[2], v3 = hp[3];
  float s = v0 + v1 + v2 + v3;
  float s2 = v0 * v0 + v1 * v1 + v2 * v2 + v3 * v3;
#pragma unroll
  for (int o = 16; o > 0; o >>= 1) {
    s  += __shfl_xor(s, o, 32);
    s2 += __shfl_xor(s2, o, 32);
  }
  float mu = s * (1.f / 128.f);
  float var = s2 * (1.f / 128.f) - mu * mu;
  float inv = rsqrtf(var + 1e-5f);
  float ns = nsrc[row];
  float* op = x_ln + (size_t)row * H_DIM + lane * 4;
  const float* gp = g + lane * 4;
  const float* bp = b + lane * 4;
  op[0] = ((v0 - mu) * inv * gp[0] + bp[0]) * ns;
  op[1] = ((v1 - mu) * inv * gp[1] + bp[1]) * ns;
  op[2] = ((v2 - mu) * inv * gp[2] + bp[2]) * ns;
  op[3] = ((v3 - mu) * inv * gp[3] + bp[3]) * ns;
}

// One block per edge, 128 lanes = one feature row; scatter-add into agg.
__global__ void scatter_kernel(const int* __restrict__ src, const int* __restrict__ dst,
                               const float* __restrict__ x_ln, float* __restrict__ agg) {
  int e = blockIdx.x;
  int c = threadIdx.x;
  int s = src[e], d = dst[e];
  atomicAdd(&agg[(size_t)d * H_DIM + c], x_ln[(size_t)s * H_DIM + c]);
}

// ---------------------------------------------------------------------------
// GraphConv GEMM (agg*norm_dst @ Wg + bg) -> LDS f16 -> 4 Gaussian heads.
// Output section order: 0=zIG_mu 1=zIG_std 2=zIA_mu 3=zIA_std (head order is
// zia_mu, zia_std, zig_mu, zig_std -> sec = (hd+2)&3).
// ---------------------------------------------------------------------------
__global__ __launch_bounds__(256)
void gconv_heads_kernel(const float* __restrict__ agg, const float* __restrict__ ndst,
                        const _Float16* __restrict__ wg_pk, const float* __restrict__ bg,
                        const _Float16* __restrict__ whead_pk,
                        const float* __restrict__ b_zia_mu, const float* __restrict__ b_zia_std,
                        const float* __restrict__ b_zig_mu, const float* __restrict__ b_zig_std,
                        float* __restrict__ out, int Nn) {
  __shared__ _Float16 lds_x[128 * H_DIM];  // 32 KB
  const int lane = threadIdx.x & 31;
  const int wave = threadIdx.x >> 5;
  const int R = wave * 16;
  const size_t nbase = (size_t)blockIdx.x * 128;
  const int m = lane & 15;
  const int hsel = lane >> 4;
  size_t grow = nbase + R + m;
  if (grow >= (size_t)Nn) grow = (size_t)Nn - 1;
  const float nd = ndst[grow];

  v16h a[4];
#pragma unroll
  for (int c = 0; c < 4; ++c)
    a[c] = loadA_g32(agg + grow * H_DIM + c * 32 + hsel * 8, nd);

  for (int j = 0; j < 8; ++j) {
    v8f acc = {};
#pragma unroll
    for (int c = 0; c < 4; ++c)
      acc = wmma16(a[c], loadB_g(wg_pk, c, j, 8, lane), acc);
    const int col = j * 16 + m;
    const float bias = bg[col];
#pragma unroll
    for (int r = 0; r < 8; ++r)
      lds_x[(R + r + hsel * 8) * H_DIM + col] = (_Float16)(acc[r] + bias);
  }

  // Rows are wave-private; same-wave DS ops are in order -> no barrier needed.
  v16h xa[4];
#pragma unroll
  for (int c = 0; c < 4; ++c)
    xa[c] = loadA_lds(&lds_x[(R + m) * H_DIM + c * 32 + hsel * 8]);

  const float* bh[4] = { b_zia_mu, b_zia_std, b_zig_mu, b_zig_std };
  for (int hd = 0; hd < 4; ++hd) {
    const _Float16* wp = whead_pk + (size_t)hd * 4 * 4 * 512;
    const int sec = (hd + 2) & 3;
    float* op = out + (size_t)sec * Nn * I_DIM;
    const int soft = hd & 1;
    for (int jt = 0; jt < 4; ++jt) {
      v8f acc = {};
#pragma unroll
      for (int c = 0; c < 4; ++c)
        acc = wmma16(xa[c], loadB_g(wp, c, jt, 4, lane), acc);
      const int col = jt * 16 + m;
      const float bias = bh[hd][col];
#pragma unroll
      for (int r = 0; r < 8; ++r) {
        size_t gr = nbase + R + r + hsel * 8;
        if (gr < (size_t)Nn) {
          float v = acc[r] + bias;
          if (soft) v = (v > 20.f) ? v : log1pf(__expf(v));
          op[gr * I_DIM + col] = v;
        }
      }
    }
  }
}

// ---------------------------------------------------------------------------
// Host side
// ---------------------------------------------------------------------------
extern "C" void kernel_launch(void* const* d_in, const int* in_sizes, int n_in,
                              void* d_out, int out_size, void* d_ws, size_t ws_size,
                              hipStream_t stream) {
  (void)n_in; (void)out_size; (void)ws_size;
  const float* actions   = (const float*)d_in[0];
  const float* hidden    = (const float*)d_in[1];
  const int*   src       = (const int*)d_in[2];
  const int*   dst       = (const int*)d_in[3];
  const float* W_ih      = (const float*)d_in[4];
  const float* W_hh      = (const float*)d_in[5];
  const float* b_ih      = (const float*)d_in[6];
  const float* b_hh      = (const float*)d_in[7];
  const float* ln_g      = (const float*)d_in[8];
  const float* ln_b      = (const float*)d_in[9];
  const float* Wg        = (const float*)d_in[10];
  const float* bg        = (const float*)d_in[11];
  const float* W_zia_mu  = (const float*)d_in[12];
  const float* b_zia_mu  = (const float*)d_in[13];
  const float* W_zia_std = (const float*)d_in[14];
  const float* b_zia_std = (const float*)d_in[15];
  const float* W_zig_mu  = (const float*)d_in[16];
  const float* b_zig_mu  = (const float*)d_in[17];
  const float* W_zig_std = (const float*)d_in[18];
  const float* b_zig_std = (const float*)d_in[19];

  const int Nn = in_sizes[1] / H_DIM;  // 50000
  const int E  = in_sizes[2];          // 800000

  // Workspace layout (~52 MB): packed weights + norms + x_ln + agg.
  size_t off = 0;
  char* ws = (char*)d_ws;
  auto alloc = [&](size_t bytes) {
    size_t o = off;
    off += (bytes + 255) & ~(size_t)255;
    return o;
  };
  _Float16* wih_pk   = (_Float16*)(ws + alloc((size_t)1 * 24 * 512 * sizeof(_Float16)));
  _Float16* whh_pk   = (_Float16*)(ws + alloc((size_t)4 * 24 * 512 * sizeof(_Float16)));
  _Float16* wg_pk    = (_Float16*)(ws + alloc((size_t)4 * 8 * 512 * sizeof(_Float16)));
  _Float16* whead_pk = (_Float16*)(ws + alloc((size_t)4 * 4 * 4 * 512 * sizeof(_Float16)));
  float* norm_src = (float*)(ws + alloc((size_t)Nn * sizeof(float)));
  float* norm_dst = (float*)(ws + alloc((size_t)Nn * sizeof(float)));
  float* x_ln     = (float*)(ws + alloc((size_t)Nn * H_DIM * sizeof(float)));
  float* agg      = (float*)(ws + alloc((size_t)Nn * H_DIM * sizeof(float)));

  float* out = (float*)d_out;
  float* h_out = out + (size_t)4 * Nn * I_DIM;  // final output: rnn state h

  // 1) Pack weights into WMMA B-operand layout (f16).
  pack_b_kernel<<<(1 * 24 * 512 + 255) / 256, 256, 0, stream>>>(W_ih, wih_pk, 1, 24, 16, 384, 1, 16);
  pack_b_kernel<<<(4 * 24 * 512 + 255) / 256, 256, 0, stream>>>(W_hh, whh_pk, 4, 24, 128, 384, 1, 128);
  pack_b_kernel<<<(4 * 8 * 512 + 255) / 256, 256, 0, stream>>>(Wg, wg_pk, 4, 8, 128, 128, 0, 128);
  const float* Wheads[4] = { W_zia_mu, W_zia_std, W_zig_mu, W_zig_std };
  for (int h = 0; h < 4; ++h)
    pack_b_kernel<<<(4 * 4 * 512 + 255) / 256, 256, 0, stream>>>(
        Wheads[h], whead_pk + (size_t)h * 4 * 4 * 512, 4, 4, 128, 64, 1, 128);

  // 2) Zero accumulators (deterministic per call; harness does not re-poison).
  zero_kernel<<<(Nn + 255) / 256, 256, 0, stream>>>(norm_src, Nn);
  zero_kernel<<<(Nn + 255) / 256, 256, 0, stream>>>(norm_dst, Nn);
  zero_kernel<<<(Nn * H_DIM + 255) / 256, 256, 0, stream>>>(agg, Nn * H_DIM);

  // 3) GRU (writes h to its output section).
  gru_kernel<<<(Nn + 127) / 128, 256, 0, stream>>>(actions, hidden, wih_pk, whh_pk,
                                                   b_ih, b_hh, h_out, Nn);

  // 4) Degrees -> symmetric norms.
  deg_kernel<<<(E + 255) / 256, 256, 0, stream>>>(src, dst, norm_src, norm_dst, E);
  norm_kernel<<<(Nn + 255) / 256, 256, 0, stream>>>(norm_src, Nn);
  norm_kernel<<<(Nn + 255) / 256, 256, 0, stream>>>(norm_dst, Nn);

  // 5) LayerNorm (+fold src-side norm).
  ln_kernel<<<(Nn + 7) / 8, 256, 0, stream>>>(h_out, ln_g, ln_b, norm_src, x_ln, Nn);

  // 6) Edge gather/scatter-add.
  scatter_kernel<<<E, 128, 0, stream>>>(src, dst, x_ln, agg);

  // 7) GraphConv GEMM + 4 Gaussian heads (WMMA).
  gconv_heads_kernel<<<(Nn + 127) / 128, 256, 0, stream>>>(
      agg, norm_dst, wg_pk, bg, whead_pk,
      b_zia_mu, b_zia_std, b_zig_mu, b_zig_std, out, Nn);
}
